// MultiScaleRetention_60258391163352
// MI455X (gfx1250) — compile-verified, matrix operand order checked by reading, not run on hardware
//
#include <hip/hip_runtime.h>

// ---------------------------------------------------------------------------
// Types / helpers
// ---------------------------------------------------------------------------
typedef unsigned short u16;
typedef __attribute__((ext_vector_type(16))) __bf16 v16bf;
typedef __attribute__((ext_vector_type(8)))  __bf16 v8bf;
typedef __attribute__((ext_vector_type(8)))  float  v8f;

#define BATCH   2
#define SEQ     4096
#define DM      1024
#define HEADS   4
#define DK      256
#define DV      512
#define VDIM    2048
#define CH      64
#define NCHUNK  (SEQ / CH)
#define DVS     64
#define SLICES  (DV / DVS)
#define MROWS   (BATCH * SEQ)   // 8192

__device__ __forceinline__ u16 f2b(float f) {
  unsigned int u = __float_as_uint(f);
  unsigned int r = u + 0x7FFFu + ((u >> 16) & 1u);   // round-to-nearest-even
  return (u16)(r >> 16);
}

// Build one 16-element bf16 WMMA fragment from two contiguous 16B groups.
// CDNA5 16-bit A/B layout: elements 0..7 at K = khalf*8+(0..7),
// elements 8..15 at K = 16 + khalf*8 + (0..7).
__device__ __forceinline__ v16bf mk_frag(const u16* p0, const u16* p1) {
  v8bf lo = *(const v8bf*)p0;
  v8bf hi = *(const v8bf*)p1;
  return __builtin_shufflevector(lo, hi, 0,1,2,3,4,5,6,7,8,9,10,11,12,13,14,15);
}

__device__ __forceinline__ v8f v8f_zero() {
  v8f z = {0.f,0.f,0.f,0.f,0.f,0.f,0.f,0.f};
  return z;
}

__device__ __forceinline__ v8f wmma_bf16(v16bf a, v16bf b, v8f c) {
  return __builtin_amdgcn_wmma_f32_16x16x32_bf16(false, a, false, b, (short)0, c,
                                                 false, false);
}

// Async global->LDS copy of 16 bytes per lane (CDNA5 ASYNCcnt path).
// LDS address = low 32 bits of the generic pointer (flat->local truncation).
__device__ __forceinline__ void async_cp16(u16* lds_dst, const u16* gsrc) {
  unsigned l = (unsigned)(size_t)lds_dst;
  asm volatile("global_load_async_to_lds_b128 %0, %1, off"
               :: "v"(l), "v"(gsrc)
               : "memory");
}

__device__ __forceinline__ void wait_async0() {
  asm volatile("s_wait_asynccnt 0x0" ::: "memory");
}

// ---------------------------------------------------------------------------
// fp32 -> bf16 conversion (grid-stride)
// ---------------------------------------------------------------------------
__global__ void f32_to_bf16_kernel(const float* __restrict__ src,
                                   u16* __restrict__ dst, size_t n) {
  size_t i = (size_t)blockIdx.x * blockDim.x + threadIdx.x;
  size_t stride = (size_t)gridDim.x * blockDim.x;
  for (; i < n; i += stride) dst[i] = f2b(src[i]);
}

// ---------------------------------------------------------------------------
// Weight transpose + bf16 convert:  src (K x N, f32)  ->  dst (N x K, bf16).
// One-time pass so GEMM B tiles become contiguous row copies (async-friendly).
// ---------------------------------------------------------------------------
__global__ __launch_bounds__(256) void transpose_bf16_kernel(
    const float* __restrict__ src, u16* __restrict__ dst, int K, int N) {
  __shared__ float tile[32][33];
  int kx = blockIdx.y * 32;
  int nx = blockIdx.x * 32;
  int tx = threadIdx.x;          // 0..31
  int ty = threadIdx.y;          // 0..7
  for (int i = 0; i < 32; i += 8)
    tile[ty + i][tx] = src[(size_t)(kx + ty + i) * N + nx + tx];
  __syncthreads();
  for (int i = 0; i < 32; i += 8)
    dst[(size_t)(nx + ty + i) * K + kx + tx] = f2b(tile[tx][ty + i]);
}

// ---------------------------------------------------------------------------
// bf16 NT GEMM:  C(MxN, f32) = A(MxK, row) * Bt(NxK, row)^T.
// Block 256 threads = 8 waves, tile 128x128x32, double-buffered LDS staged
// with GLOBAL_LOAD_ASYNC_TO_LDS_B128 (ASYNCcnt), WMMA consumes via b128 DS
// loads.  Waves arranged 2x4; each wave owns a 64x32 output region.
// ---------------------------------------------------------------------------
#define GTM 128
#define GTN 128
#define GTK 32
#define GPAD 8

__global__ __launch_bounds__(256) void gemm_bf16_nt(const u16* __restrict__ A,
                                                    const u16* __restrict__ Bt,
                                                    float* __restrict__ C,
                                                    int M, int N, int K) {
  __shared__ __align__(16) u16 As[2][GTM][GTK + GPAD];
  __shared__ __align__(16) u16 Bs[2][GTN][GTK + GPAD];
  const int tid  = threadIdx.x;
  const int w    = tid >> 5, lane = tid & 31;
  const int lrow = lane & 15, lhi = lane >> 4;
  const int wM   = w >> 2, wN = w & 3;
  const int bM   = blockIdx.y * GTM, bN = blockIdx.x * GTN;

  const int sr = tid >> 1;            // staged row 0..127
  const int sc = (tid & 1) << 4;      // k-group 0 or 16

  v8f acc[4][2];
  for (int mt = 0; mt < 4; ++mt)
    for (int nt = 0; nt < 2; ++nt) acc[mt][nt] = v8f_zero();

  // prologue: stage k0 = 0 into buffer 0
  {
    const u16* ga = A  + (size_t)(bM + sr) * K + sc;
    const u16* gb = Bt + (size_t)(bN + sr) * K + sc;
    async_cp16(&As[0][sr][sc],     ga);
    async_cp16(&As[0][sr][sc + 8], ga + 8);
    async_cp16(&Bs[0][sr][sc],     gb);
    async_cp16(&Bs[0][sr][sc + 8], gb + 8);
  }

  int buf = 0;
  for (int k0 = 0; k0 < K; k0 += GTK) {
    wait_async0();        // this wave's async copies have landed in LDS
    __syncthreads();      // all waves' copies visible

    if (k0 + GTK < K) {   // stage next tile into the other buffer (overlaps WMMA)
      int nb = buf ^ 1;
      const u16* ga = A  + (size_t)(bM + sr) * K + (k0 + GTK) + sc;
      const u16* gb = Bt + (size_t)(bN + sr) * K + (k0 + GTK) + sc;
      async_cp16(&As[nb][sr][sc],     ga);
      async_cp16(&As[nb][sr][sc + 8], ga + 8);
      async_cp16(&Bs[nb][sr][sc],     gb);
      async_cp16(&Bs[nb][sr][sc + 8], gb + 8);
    }

    v16bf bfr[2];
    for (int nt = 0; nt < 2; ++nt) {
      int n = wN * 32 + nt * 16 + lrow;
      bfr[nt] = mk_frag(&Bs[buf][n][lhi * 8], &Bs[buf][n][16 + lhi * 8]);
    }
    for (int mt = 0; mt < 4; ++mt) {
      int m = wM * 64 + mt * 16 + lrow;
      v16bf afr = mk_frag(&As[buf][m][lhi * 8], &As[buf][m][16 + lhi * 8]);
      for (int nt = 0; nt < 2; ++nt)
        acc[mt][nt] = wmma_bf16(afr, bfr[nt], acc[mt][nt]);
    }
    buf ^= 1;
  }

  // D layout: element r -> M = lhi*8 + r, N = lrow
  for (int mt = 0; mt < 4; ++mt)
    for (int nt = 0; nt < 2; ++nt)
      for (int r = 0; r < 8; ++r) {
        int row = bM + wM * 64 + mt * 16 + lhi * 8 + r;
        int col = bN + wN * 32 + nt * 16 + lrow;
        C[(size_t)row * N + col] = acc[mt][nt][r];
      }
}

// ---------------------------------------------------------------------------
// Rotary + scale folding:  q2 = rot(q) * gamma^(t%64+1) / 16
//                          k2 = rot(k) * gamma^(63 - t%64)
// Layout (B, N, H, DK) bf16.
// ---------------------------------------------------------------------------
__global__ __launch_bounds__(256) void rotary_kernel(const float* __restrict__ qf,
                                                     const float* __restrict__ kf,
                                                     u16* __restrict__ q2,
                                                     u16* __restrict__ k2) {
  size_t idx = (size_t)blockIdx.x * 256 + threadIdx.x;
  if (idx >= (size_t)MROWS * HEADS * (DK / 2)) return;
  int i = (int)(idx & 127);
  size_t rest = idx >> 7;
  int h = (int)(rest & (HEADS - 1));
  size_t bn = rest >> 2;                  // b*SEQ + t
  int t = (int)(bn & (SEQ - 1));

  float invf = exp2f(-((float)i / 128.0f) * 13.287712379549449f);  // 10000^(-2i/256)
  float fr = (float)t * invf;
  float sn, cs;
  __sincosf(fr, &sn, &cs);

  size_t base = (bn * HEADS + h) * DK;
  float qa0 = qf[base + i],       qb0 = qf[base + 128 + i];
  float ka0 = kf[base + i],       kb0 = kf[base + 128 + i];
  float qa = qa0 * cs - qb0 * sn, qb = qb0 * cs + qa0 * sn;
  float ka = ka0 * cs - kb0 * sn, kb = kb0 * cs + ka0 * sn;

  float gamma = 1.0f - exp2f(-5.0f - (float)h);
  int tc = t & (CH - 1);
  float cq = powf(gamma, (float)(tc + 1)) * 0.0625f;   // fold dk^-0.5 = 1/16
  float ck = powf(gamma, (float)(CH - 1 - tc));

  q2[base + i]       = f2b(qa * cq);
  q2[base + 128 + i] = f2b(qb * cq);
  k2[base + i]       = f2b(ka * ck);
  k2[base + 128 + i] = f2b(kb * ck);
}

// ---------------------------------------------------------------------------
// Chunked retention scan.  Grid: B*H*SLICES = 64 blocks, 256 threads (8 waves).
// Each block owns a 64-wide dv slice of the state S (256 x 64), kept in
// registers (8 v8f accumulators per lane per wave).  All einsums via WMMA.
// ---------------------------------------------------------------------------
struct RetSmem {
  union {
    u16 sT[DVS][DK + 8];   // S^T in bf16: [n][d]   (phase: o_inter)
    u16 kT[DK][CH + 8];    // k2^T: [d][t]          (phase: S update)
  } u;
  u16 vT[DVS][CH + 8];     // v^T: [n][t]
  u16 at[CH][CH + 8];      // decayed attention [i][j] in bf16
};

__global__ __launch_bounds__(256) void retention_kernel(const u16* __restrict__ q2,
                                                        const u16* __restrict__ k2,
                                                        const u16* __restrict__ vb,
                                                        float* __restrict__ of) {
  __shared__ __align__(16) RetSmem sm;
  const int blk   = blockIdx.x;
  const int slice = blk & (SLICES - 1);
  const int h     = (blk >> 3) & (HEADS - 1);
  const int b     = blk >> 5;
  const int tid   = threadIdx.x;
  const int w     = tid >> 5, lane = tid & 31;
  const int lrow  = lane & 15, lhi = lane >> 4;

  const float gamma = 1.0f - exp2f(-5.0f - (float)h);
  const float g64   = powf(gamma, 64.0f);
  const float inv64 = 1.0f / g64;

  v8f S[8];
  for (int i = 0; i < 8; ++i) S[i] = v8f_zero();

  for (int c = 0; c < NCHUNK; ++c) {
    const int t0 = c * CH;

    // --- stage v^T slice:  vT[n][t] = v[b, t0+t, h, slice*64+n]
    for (int it = 0; it < (DVS * CH) / 256; ++it) {
      int idx = it * 256 + tid;
      int n = idx & (DVS - 1);
      int t = idx >> 6;
      sm.vT[n][t] = vb[((size_t)(b * SEQ + t0 + t) * HEADS + h) * DV + slice * DVS + n];
    }

    // --- dump register S to bf16 S^T  (D layout: M = lhi*8+r, N = lrow)
    for (int i = 0; i < 8; ++i) {
      int ti = w * 8 + i;
      int mT = ti >> 2, nT = ti & 3;
      int n = nT * 16 + lrow;
      int dbase = mT * 16 + lhi * 8;
      for (int r = 0; r < 8; ++r) sm.u.sT[n][dbase + r] = f2b(S[i][r]);
    }

    // --- decayed attention: (q2 @ k2^T) * gamma^-64, causal-masked
    v8f zt[2];
    int iT[2], jT[2];
    for (int s = 0; s < 2; ++s) {
      int atile = w * 2 + s;
      iT[s] = atile >> 2; jT[s] = atile & 3;
      zt[s] = v8f_zero();
    }
    for (int kk = 0; kk < DK; kk += 32) {
      for (int s = 0; s < 2; ++s) {
        int ti_ = t0 + iT[s] * 16 + lrow;
        const u16* arow = q2 + ((size_t)(b * SEQ + ti_) * HEADS + h) * DK;
        v16bf a = mk_frag(arow + kk + lhi * 8, arow + kk + 16 + lhi * 8);
        int tj = t0 + jT[s] * 16 + lrow;
        const u16* brow = k2 + ((size_t)(b * SEQ + tj) * HEADS + h) * DK;
        v16bf bb = mk_frag(brow + kk + lhi * 8, brow + kk + 16 + lhi * 8);
        zt[s] = wmma_bf16(a, bb, zt[s]);
      }
    }
    for (int s = 0; s < 2; ++s)
      for (int r = 0; r < 8; ++r) {
        int i = iT[s] * 16 + lhi * 8 + r;
        int j = jT[s] * 16 + lrow;
        float v = (i >= j) ? zt[s][r] * inv64 : 0.0f;
        sm.at[i][j] = f2b(v);
      }
    __syncthreads();

    // --- o = q2 @ S  +  attn @ v   (64 x 64 slice)
    for (int s = 0; s < 2; ++s) {
      int ot = w * 2 + s;
      int oiT = ot >> 2, onT = ot & 3;
      v8f o = v8f_zero();
      int ti_ = t0 + oiT * 16 + lrow;
      const u16* arow = q2 + ((size_t)(b * SEQ + ti_) * HEADS + h) * DK;
      int n = onT * 16 + lrow;
      for (int kk = 0; kk < DK; kk += 32) {
        v16bf a  = mk_frag(arow + kk + lhi * 8, arow + kk + 16 + lhi * 8);
        v16bf bb = mk_frag(&sm.u.sT[n][kk + lhi * 8], &sm.u.sT[n][kk + 16 + lhi * 8]);
        o = wmma_bf16(a, bb, o);
      }
      int irow = oiT * 16 + lrow;
      for (int kk = 0; kk < CH; kk += 32) {
        v16bf a  = mk_frag(&sm.at[irow][kk + lhi * 8], &sm.at[irow][kk + 16 + lhi * 8]);
        v16bf bb = mk_frag(&sm.vT[n][kk + lhi * 8], &sm.vT[n][kk + 16 + lhi * 8]);
        o = wmma_bf16(a, bb, o);
      }
      for (int r = 0; r < 8; ++r) {
        int tg = t0 + oiT * 16 + lhi * 8 + r;
        int ng = slice * DVS + onT * 16 + lrow;
        of[((size_t)(b * SEQ + tg) * HEADS + h) * DV + ng] = o[r];
      }
    }
    __syncthreads();   // sT fully consumed; safe to overwrite with kT

    // --- stage k2^T:  kT[d][t] = k2[b, t0+t, h, d]
    for (int it = 0; it < (DK * CH) / 256; ++it) {
      int idx = it * 256 + tid;
      int d = idx & (DK - 1);
      int t = idx >> 8;
      sm.u.kT[d][t] = k2[((size_t)(b * SEQ + t0 + t) * HEADS + h) * DK + d];
    }
    __syncthreads();

    // --- S = gamma^64 * S + k2^T @ v
    for (int i = 0; i < 8; ++i) {
      int ti = w * 8 + i;
      int mT = ti >> 2, nT = ti & 3;
      for (int r = 0; r < 8; ++r) S[i][r] *= g64;
      int d = mT * 16 + lrow;
      int n = nT * 16 + lrow;
      for (int kk = 0; kk < CH; kk += 32) {
        v16bf a  = mk_frag(&sm.u.kT[d][kk + lhi * 8], &sm.u.kT[d][kk + 16 + lhi * 8]);
        v16bf bb = mk_frag(&sm.vT[n][kk + lhi * 8], &sm.vT[n][kk + 16 + lhi * 8]);
        S[i] = wmma_bf16(a, bb, S[i]);
      }
    }
    __syncthreads();   // kT/vT consumed before next chunk restages
  }
}

// ---------------------------------------------------------------------------
// RMSNorm(head) * silu(gate) -> bf16.  One block per (b,n,h), 128 threads.
// ---------------------------------------------------------------------------
__global__ __launch_bounds__(128) void norm_gate_kernel(const float* __restrict__ o,
                                                        const float* __restrict__ g,
                                                        const float* __restrict__ gw,
                                                        u16* __restrict__ y) {
  __shared__ float red[128];
  int blk = blockIdx.x;
  int h   = blk & (HEADS - 1);
  int bn  = blk >> 2;
  int tid = threadIdx.x;
  const float* orow = o + ((size_t)bn * HEADS + h) * DV;

  float4 xv = *(const float4*)(orow + tid * 4);
  red[tid] = xv.x * xv.x + xv.y * xv.y + xv.z * xv.z + xv.w * xv.w;
  __syncthreads();
  for (int st = 64; st > 0; st >>= 1) {
    if (tid < st) red[tid] += red[tid + st];
    __syncthreads();
  }
  float inv = rsqrtf(red[0] / (float)DV + 1e-5f);

  size_t gbase = (size_t)bn * VDIM + (size_t)h * DV + tid * 4;
  float4 gv = *(const float4*)(g + gbase);
  float4 wv = *(const float4*)(gw + tid * 4);
  float s0 = gv.x / (1.0f + __expf(-gv.x));
  float s1 = gv.y / (1.0f + __expf(-gv.y));
  float s2 = gv.z / (1.0f + __expf(-gv.z));
  float s3 = gv.w / (1.0f + __expf(-gv.w));
  y[gbase + 0] = f2b(xv.x * inv * wv.x * s0);
  y[gbase + 1] = f2b(xv.y * inv * wv.y * s1);
  y[gbase + 2] = f2b(xv.z * inv * wv.z * s2);
  y[gbase + 3] = f2b(xv.w * inv * wv.w * s3);
}

// ---------------------------------------------------------------------------
// Launcher
// ---------------------------------------------------------------------------
extern "C" void kernel_launch(void* const* d_in, const int* in_sizes, int n_in,
                              void* d_out, int out_size, void* d_ws, size_t ws_size,
                              hipStream_t stream) {
  (void)in_sizes; (void)n_in; (void)out_size; (void)ws_size;
  const float* x   = (const float*)d_in[0];
  const float* Wq  = (const float*)d_in[1];
  const float* Wk  = (const float*)d_in[2];
  const float* Wv  = (const float*)d_in[3];
  const float* Wg  = (const float*)d_in[4];
  const float* Wo  = (const float*)d_in[5];
  const float* gnw = (const float*)d_in[6];
  float* out = (float*)d_out;

  char* ws = (char*)d_ws;
  size_t off = 0;
  auto take = [&](size_t bytes) -> char* {
    char* p = ws + off;
    off += (bytes + 255) & ~(size_t)255;
    return p;
  };
  u16*   xb  = (u16*)  take((size_t)MROWS * DM * 2);
  u16*   Wqt = (u16*)  take((size_t)DM * DM * 2);      // N x K bf16
  u16*   Wkt = (u16*)  take((size_t)DM * DM * 2);
  u16*   Wvt = (u16*)  take((size_t)VDIM * DM * 2);
  u16*   Wgt = (u16*)  take((size_t)VDIM * DM * 2);
  u16*   Wot = (u16*)  take((size_t)DM * VDIM * 2);
  float* qf  = (float*)take((size_t)MROWS * DM * 4);    // later reused as y (bf16)
  float* kf  = (float*)take((size_t)MROWS * DM * 4);
  float* vf  = (float*)take((size_t)MROWS * VDIM * 4);  // later reused as o (f32)
  float* gf  = (float*)take((size_t)MROWS * VDIM * 4);
  u16*   q2b = (u16*)  take((size_t)MROWS * DM * 2);
  u16*   k2b = (u16*)  take((size_t)MROWS * DM * 2);
  u16*   vb  = (u16*)  take((size_t)MROWS * VDIM * 2);
  u16*   yb  = (u16*)qf;    // alias: qf dead after rotary
  float* of  = vf;          // alias: vf dead after bf16 conversion

  // activations -> bf16; weights -> transposed bf16 (NxK)
  f32_to_bf16_kernel<<<4096, 256, 0, stream>>>(x, xb, (size_t)MROWS * DM);
  transpose_bf16_kernel<<<dim3(DM / 32, DM / 32), dim3(32, 8), 0, stream>>>(Wq, Wqt, DM, DM);
  transpose_bf16_kernel<<<dim3(DM / 32, DM / 32), dim3(32, 8), 0, stream>>>(Wk, Wkt, DM, DM);
  transpose_bf16_kernel<<<dim3(VDIM / 32, DM / 32), dim3(32, 8), 0, stream>>>(Wv, Wvt, DM, VDIM);
  transpose_bf16_kernel<<<dim3(VDIM / 32, DM / 32), dim3(32, 8), 0, stream>>>(Wg, Wgt, DM, VDIM);
  transpose_bf16_kernel<<<dim3(DM / 32, VDIM / 32), dim3(32, 8), 0, stream>>>(Wo, Wot, VDIM, DM);

  // projections (async double-buffered WMMA GEMMs)
  gemm_bf16_nt<<<dim3(DM / 128, MROWS / 128), 256, 0, stream>>>(xb, Wqt, qf, MROWS, DM, DM);
  gemm_bf16_nt<<<dim3(DM / 128, MROWS / 128), 256, 0, stream>>>(xb, Wkt, kf, MROWS, DM, DM);
  gemm_bf16_nt<<<dim3(VDIM / 128, MROWS / 128), 256, 0, stream>>>(xb, Wvt, vf, MROWS, VDIM, DM);
  gemm_bf16_nt<<<dim3(VDIM / 128, MROWS / 128), 256, 0, stream>>>(xb, Wgt, gf, MROWS, VDIM, DM);

  f32_to_bf16_kernel<<<8192, 256, 0, stream>>>(vf, vb, (size_t)MROWS * VDIM);

  // rotary + decay folding -> q2/k2 bf16
  rotary_kernel<<<(MROWS * HEADS * (DK / 2)) / 256, 256, 0, stream>>>(qf, kf, q2b, k2b);

  // chunked retention scan
  retention_kernel<<<BATCH * HEADS * SLICES, 256, 0, stream>>>(q2b, k2b, vb, of);

  // rmsnorm * silu(gate) -> bf16
  norm_gate_kernel<<<MROWS * HEADS, 128, 0, stream>>>(of, gf, gnw, yb);

  // output projection
  gemm_bf16_nt<<<dim3(DM / 128, MROWS / 128), 256, 0, stream>>>(yb, Wot, out, MROWS, DM, VDIM);
}
